// Decoder_5317169512676
// MI455X (gfx1250) — compile-verified, matrix operand order checked by reading, not run on hardware
//
#include <hip/hip_runtime.h>

// Problem constants (match reference)
#define B_ 2
#define L_ 4096
#define D_ 1024
#define NL_ 2
#define M_ (B_ * L_)        // 8192 rows
#define CHUNK 64
#define NCHUNK (L_ / CHUNK) // 64 chunks per batch row

typedef __attribute__((ext_vector_type(16))) __bf16 v16bf;
typedef __attribute__((ext_vector_type(8)))  __bf16 v8bf;
typedef __attribute__((ext_vector_type(8)))  float  v8f;

__device__ __forceinline__ unsigned short f2bf(float x) {
    unsigned u = __float_as_uint(x);
    unsigned r = u + 0x7FFFu + ((u >> 16) & 1u);   // round-to-nearest-even
    return (unsigned short)(r >> 16);
}

// ---------------------------------------------------------------------------
// RMS norm: one block per row; writes f32 copy (for the scan) + bf16 copy
// (WMMA A-matrix operand).
// ---------------------------------------------------------------------------
__global__ __launch_bounds__(256) void k_rmsnorm(const float* __restrict__ src,
                                                 float* __restrict__ hn,
                                                 unsigned short* __restrict__ hnb) {
    const int row = blockIdx.x;
    const int t = threadIdx.x;
    const float4 v = reinterpret_cast<const float4*>(src + (size_t)row * D_)[t];
    float ss = v.x * v.x + v.y * v.y + v.z * v.z + v.w * v.w;
#pragma unroll
    for (int off = 16; off > 0; off >>= 1) ss += __shfl_down(ss, off, 32);
    __shared__ float wsum[8];
    __shared__ float rbc;
    const int lane = t & 31, w = t >> 5;
    if (lane == 0) wsum[w] = ss;
    __syncthreads();
    if (t == 0) {
        float tot = 0.0f;
#pragma unroll
        for (int i = 0; i < 8; ++i) tot += wsum[i];
        rbc = rsqrtf(tot * (1.0f / (float)D_) + 1.1920929e-07f);
    }
    __syncthreads();
    const float r = rbc;
    float4 o;
    o.x = v.x * r; o.y = v.y * r; o.z = v.z * r; o.w = v.w * r;
    reinterpret_cast<float4*>(hn + (size_t)row * D_)[t] = o;
    uint2 pk;
    pk.x = (unsigned)f2bf(o.x) | ((unsigned)f2bf(o.y) << 16);
    pk.y = (unsigned)f2bf(o.z) | ((unsigned)f2bf(o.w) << 16);
    reinterpret_cast<uint2*>(hnb + (size_t)row * D_)[t] = pk;
}

// f32 -> bf16 (weights), raw-bit storage
__global__ __launch_bounds__(256) void k_f32_to_bf16(const float* __restrict__ x,
                                                     unsigned short* __restrict__ y,
                                                     int n) {
    int i = blockIdx.x * 256 + threadIdx.x;
    if (i < n) y[i] = f2bf(x[i]);
}

// ---------------------------------------------------------------------------
// Q/K projection GEMM: out[m,n] = sum_d A[m,d]*W[n,d] + bias[n]
// Block tile 128x128, 8 waves (4x2), wave tile 32x64, K-step 32 through LDS.
// blockIdx.z: 0 -> (Wq,bq,Q), 1 -> (Wk,bk,K).
// ---------------------------------------------------------------------------
#define LDST 40   // padded LDS row stride (half-words): 80B rows, conflict-free b128
__global__ __launch_bounds__(256) void k_qk_gemm(
        const unsigned short* __restrict__ Abf,
        const unsigned short* __restrict__ WqB,
        const unsigned short* __restrict__ WkB,
        const float* __restrict__ bq, const float* __restrict__ bk,
        float* __restrict__ Qo, float* __restrict__ Ko) {
    const unsigned short* Wbf  = blockIdx.z ? WkB : WqB;
    const float*          bias = blockIdx.z ? bk  : bq;
    float*                out  = blockIdx.z ? Ko  : Qo;

    __shared__ __align__(16) unsigned short shA[128 * LDST];
    __shared__ __align__(16) unsigned short shB[128 * LDST];

    const int t      = threadIdx.x;
    const int lane   = t & 31;
    const int wid    = t >> 5;
    const int waveM0 = (wid & 3) * 32;
    const int waveN0 = (wid >> 2) * 64;
    const int half   = lane >> 4;   // selects K-half per ISA 16-bit layout
    const int mr     = lane & 15;   // row (A) / column (B) within 16-tile
    const int rowBlk = blockIdx.x * 128;
    const int colBlk = blockIdx.y * 128;

    v8f acc[2][4];
#pragma unroll
    for (int mi = 0; mi < 2; ++mi)
#pragma unroll
        for (int ni = 0; ni < 4; ++ni) acc[mi][ni] = {};

    for (int k0 = 0; k0 < D_; k0 += 32) {
        __syncthreads();
#pragma unroll
        for (int j = 0; j < 2; ++j) {
            int c   = t + 256 * j;     // 512 16B chunks per tile
            int row = c >> 2;
            int col = c & 3;
            const uint4 va = *reinterpret_cast<const uint4*>(
                Abf + (size_t)(rowBlk + row) * D_ + k0 + col * 8);
            *reinterpret_cast<uint4*>(&shA[row * LDST + col * 8]) = va;
            const uint4 vb = *reinterpret_cast<const uint4*>(
                Wbf + (size_t)(colBlk + row) * D_ + k0 + col * 8);
            *reinterpret_cast<uint4*>(&shB[row * LDST + col * 8]) = vb;
            if (k0 + 32 < D_) {
                __builtin_prefetch(Abf + (size_t)(rowBlk + row) * D_ + k0 + 32 + col * 8, 0, 3);
                __builtin_prefetch(Wbf + (size_t)(colBlk + row) * D_ + k0 + 32 + col * 8, 0, 3);
            }
        }
        __syncthreads();

        v16bf afr[2], bfr[4];
#pragma unroll
        for (int mi = 0; mi < 2; ++mi) {
            const unsigned short* rp = &shA[(waveM0 + mi * 16 + mr) * LDST];
            v8bf lo = *reinterpret_cast<const v8bf*>(rp + half * 8);        // K 8*half..
            v8bf hi = *reinterpret_cast<const v8bf*>(rp + 16 + half * 8);   // K 16+8*half..
#pragma unroll
            for (int e = 0; e < 8; ++e) { afr[mi][e] = lo[e]; afr[mi][e + 8] = hi[e]; }
        }
#pragma unroll
        for (int ni = 0; ni < 4; ++ni) {
            const unsigned short* rp = &shB[(waveN0 + ni * 16 + mr) * LDST];
            v8bf lo = *reinterpret_cast<const v8bf*>(rp + half * 8);
            v8bf hi = *reinterpret_cast<const v8bf*>(rp + 16 + half * 8);
#pragma unroll
            for (int e = 0; e < 8; ++e) { bfr[ni][e] = lo[e]; bfr[ni][e + 8] = hi[e]; }
        }
#pragma unroll
        for (int mi = 0; mi < 2; ++mi)
#pragma unroll
            for (int ni = 0; ni < 4; ++ni)
                acc[mi][ni] = __builtin_amdgcn_wmma_f32_16x16x32_bf16(
                    false, afr[mi], false, bfr[ni], (short)0, acc[mi][ni], false, false);
    }

    // Epilogue: C layout — lane holds N=lane&15; VGPR r -> M = (lane>>4)*8 + r
#pragma unroll
    for (int ni = 0; ni < 4; ++ni) {
        const int n  = colBlk + waveN0 + ni * 16 + mr;
        const float bv = bias[n];
#pragma unroll
        for (int mi = 0; mi < 2; ++mi) {
            const int mrow = rowBlk + waveM0 + mi * 16 + half * 8;
            float* op = out + (size_t)mrow * D_ + n;
#pragma unroll
            for (int r = 0; r < 8; ++r) op[(size_t)r * D_] = acc[mi][ni][r] + bv;
        }
    }
}

// ---------------------------------------------------------------------------
// Routing: cos(Q[b,l-1], K[b,l]) -> A -> p, boundary mask. One block per (l,b).
// ---------------------------------------------------------------------------
__global__ __launch_bounds__(128) void k_routing(const float* __restrict__ Qf,
                                                 const float* __restrict__ Kf,
                                                 const unsigned char* __restrict__ mask,
                                                 float* __restrict__ p,
                                                 int* __restrict__ bmask) {
    const int l = blockIdx.x;
    const int b = blockIdx.y;
    const int t = threadIdx.x;
    __shared__ float sred[3][4];
    float A = 1.0f;                       // A[b,0] = 1 (padded)
    if (l > 0) {
        const float* Qr = Qf + ((size_t)b * L_ + (l - 1)) * D_;
        const float* Kr = Kf + ((size_t)b * L_ + l) * D_;
        float qk = 0.0f, qq = 0.0f, kk = 0.0f;
        for (int d = t; d < D_; d += 128) {
            float q = Qr[d], k = Kr[d];
            qk += q * k; qq += q * q; kk += k * k;
        }
#pragma unroll
        for (int off = 16; off > 0; off >>= 1) {
            qk += __shfl_down(qk, off, 32);
            qq += __shfl_down(qq, off, 32);
            kk += __shfl_down(kk, off, 32);
        }
        const int lane = t & 31, w = t >> 5;
        if (lane == 0) { sred[0][w] = qk; sred[1][w] = qq; sred[2][w] = kk; }
        __syncthreads();
        if (t == 0) {
            float dqk = 0, dqq = 0, dkk = 0;
#pragma unroll
            for (int i = 0; i < 4; ++i) { dqk += sred[0][i]; dqq += sred[1][i]; dkk += sred[2][i]; }
            float nq = fmaxf(sqrtf(dqq), 1e-12f);
            float nk = fmaxf(sqrtf(dkk), 1e-12f);
            float cosv = dqk / (nq * nk);
            A = fminf(fmaxf(0.5f * (1.0f - cosv), 0.0f), 1.0f);
        }
    }
    if (t == 0) {
        const size_t o = (size_t)b * L_ + l;
        p[o] = fminf(fmaxf(A, 1e-4f), 1.0f - 1e-4f);
        int bm = (A > 0.5f) ? 1 : 0;     // argmax([1-A, A]) first-max tie -> 0
        bmask[o] = mask[o] ? bm : 0;
    }
}

// ---------------------------------------------------------------------------
// Inclusive cumsum of boundary mask -> gather index. One block per batch row.
// ---------------------------------------------------------------------------
__global__ __launch_bounds__(1024) void k_cumsum_idx(const int* __restrict__ bmask,
                                                     int* __restrict__ idx) {
    const int b = blockIdx.x;
    const int t = threadIdx.x;
    __shared__ int s[1024];
    const size_t base = (size_t)b * L_;
    int v0 = bmask[base + t * 4 + 0];
    int v1 = bmask[base + t * 4 + 1];
    int v2 = bmask[base + t * 4 + 2];
    int v3 = bmask[base + t * 4 + 3];
    s[t] = v0 + v1 + v2 + v3;
    __syncthreads();
    for (int off = 1; off < 1024; off <<= 1) {
        int add = (t >= off) ? s[t - off] : 0;
        __syncthreads();
        s[t] += add;
        __syncthreads();
    }
    int c = (t == 0) ? 0 : s[t - 1];
    c += v0; idx[base + t * 4 + 0] = min(max(c - 1, 0), L_ - 1);
    c += v1; idx[base + t * 4 + 1] = min(max(c - 1, 0), L_ - 1);
    c += v2; idx[base + t * 4 + 2] = min(max(c - 1, 0), L_ - 1);
    c += v3; idx[base + t * 4 + 3] = min(max(c - 1, 0), L_ - 1);
}

// ---------------------------------------------------------------------------
// Chunked linear-recurrence scan. mode 0 = per-chunk affine (A=prod q, B[d]);
// mode 1 = replay with carried prefix, writing the layer output.
// Fuses upsample(h) + rw*upsample(enc) via linearity (same coefficients).
// ---------------------------------------------------------------------------
__global__ __launch_bounds__(1024) void k_upsample(
        const float* __restrict__ hn, const float* __restrict__ enc,
        const float* __restrict__ p, const int* __restrict__ idx,
        const float* __restrict__ rw_all, int layer,
        const float* __restrict__ carryIn,   // mode 1: [B,NCHUNK,D]
        float* __restrict__ Bc,              // mode 0: [B,NCHUNK,D]
        float* __restrict__ Aprod,           // mode 0: [B,NCHUNK]
        float* __restrict__ outp,            // mode 1: [B,L,D]
        int mode) {
    const int s = blockIdx.x;
    const int b = blockIdx.y;
    const int d = threadIdx.x;
    const int l0 = s * CHUNK;
    __shared__ float pl[CHUNK], ql[CHUNK];
    __shared__ int jx[CHUNK];
    if (d < CHUNK) {
        const int gl = l0 + d;
        const size_t o = (size_t)b * L_ + gl;
        pl[d] = p[o];
        ql[d] = (gl == 0) ? 1.0f : 1.0f - p[o - 1];
        jx[d] = idx[o];
    }
    __syncthreads();
    const float rw = rw_all[layer];
    float u = (mode == 1) ? carryIn[((size_t)b * NCHUNK + s) * D_ + d] : 0.0f;
    for (int l = 0; l < CHUNK; ++l) {
        const size_t ro = ((size_t)b * L_ + jx[l]) * D_ + d;
        const float zc = hn[ro] + rw * enc[ro];
        u = ql[l] * u + pl[l] * zc;
        if (mode == 1) outp[((size_t)b * L_ + l0 + l) * D_ + d] = u;
    }
    if (mode == 0) {
        Bc[((size_t)b * NCHUNK + s) * D_ + d] = u;
        if (d == 0) {
            float a = 1.0f;
            for (int l = 0; l < CHUNK; ++l) a *= ql[l];
            Aprod[(size_t)b * NCHUNK + s] = a;
        }
    }
}

// Cross-chunk carry: carry into chunk s = u_end(chunk s-1). 64 steps.
__global__ __launch_bounds__(1024) void k_carry(const float* __restrict__ Bc,
                                                const float* __restrict__ Aprod,
                                                float* __restrict__ carry) {
    const int b = blockIdx.x;
    const int d = threadIdx.x;
    float c = 0.0f;
    for (int s = 0; s < NCHUNK; ++s) {
        const size_t o = ((size_t)b * NCHUNK + s) * D_ + d;
        carry[o] = c;
        c = Aprod[(size_t)b * NCHUNK + s] * c + Bc[o];
    }
}

// ---------------------------------------------------------------------------
extern "C" void kernel_launch(void* const* d_in, const int* in_sizes, int n_in,
                              void* d_out, int out_size, void* d_ws, size_t ws_size,
                              hipStream_t stream) {
    const float* hidden  = (const float*)d_in[0];
    const float* encoder = (const float*)d_in[1];
    const float* Wq      = (const float*)d_in[2];
    const float* bq      = (const float*)d_in[3];
    const float* Wk      = (const float*)d_in[4];
    const float* bk      = (const float*)d_in[5];
    const float* rw      = (const float*)d_in[6];
    const unsigned char* mask = (const unsigned char*)d_in[7];
    float* out = (float*)d_out;
    (void)in_sizes; (void)n_in; (void)out_size; (void)ws_size;

    char* ws = (char*)d_ws;
    size_t off = 0;
    auto alloc = [&](size_t bytes) -> char* {
        char* q = ws + off;
        off += (bytes + 255) & ~(size_t)255;
        return q;
    };
    float*          hnf  = (float*)         alloc((size_t)M_ * D_ * 4);
    unsigned short* hnb  = (unsigned short*)alloc((size_t)M_ * D_ * 2);
    unsigned short* wqB  = (unsigned short*)alloc((size_t)D_ * D_ * 2);
    unsigned short* wkB  = (unsigned short*)alloc((size_t)D_ * D_ * 2);
    float*          Qf   = (float*)alloc((size_t)M_ * D_ * 4);
    float*          Kf   = (float*)alloc((size_t)M_ * D_ * 4);
    float*          pbuf = (float*)alloc((size_t)M_ * 4);
    int*            bm   = (int*)  alloc((size_t)M_ * 4);
    int*            ix   = (int*)  alloc((size_t)M_ * 4);
    float*          Bc   = (float*)alloc((size_t)B_ * NCHUNK * D_ * 4);
    float*          Ap   = (float*)alloc((size_t)B_ * NCHUNK * 4);
    float*          Cin  = (float*)alloc((size_t)B_ * NCHUNK * D_ * 4);

    for (int layer = 0; layer < NL_; ++layer) {
        const float* src = (layer == 0) ? hidden : out;
        const float* enc = encoder + (size_t)(NL_ - 1 - layer) * M_ * D_;

        k_f32_to_bf16<<<dim3((D_ * D_) / 256), 256, 0, stream>>>(
            Wq + (size_t)layer * D_ * D_, wqB, D_ * D_);
        k_f32_to_bf16<<<dim3((D_ * D_) / 256), 256, 0, stream>>>(
            Wk + (size_t)layer * D_ * D_, wkB, D_ * D_);
        k_rmsnorm<<<dim3(M_), 256, 0, stream>>>(src, hnf, hnb);
        k_qk_gemm<<<dim3(M_ / 128, D_ / 128, 2), 256, 0, stream>>>(
            hnb, wqB, wkB, bq + (size_t)layer * D_, bk + (size_t)layer * D_, Qf, Kf);
        k_routing<<<dim3(L_, B_), 128, 0, stream>>>(Qf, Kf, mask, pbuf, bm);
        k_cumsum_idx<<<dim3(B_), 1024, 0, stream>>>(bm, ix);
        k_upsample<<<dim3(NCHUNK, B_), 1024, 0, stream>>>(
            hnf, enc, pbuf, ix, rw, layer, nullptr, Bc, Ap, nullptr, 0);
        k_carry<<<dim3(B_), 1024, 0, stream>>>(Bc, Ap, Cin);
        k_upsample<<<dim3(NCHUNK, B_), 1024, 0, stream>>>(
            hnf, enc, pbuf, ix, rw, layer, Cin, nullptr, nullptr, out, 1);
    }
}
